// INTP_Model_13357348290605
// MI455X (gfx1250) — matrix operand order repaired and checked by reading, not compiled, not run on hardware
//
#include <hip/hip_runtime.h>

// ---------------- problem constants (reference shapes) ----------------
constexpr int N_   = 40000;
constexpr int E_   = 320000;
constexpr int D_   = 128;
constexpr int P_   = 16;
constexpr int FIN_ = 64;
constexpr int L_   = 3;
constexpr int G_   = 40;

// ---------------- types ----------------
typedef __attribute__((ext_vector_type(16))) _Float16     v16h;
typedef __attribute__((ext_vector_type(8)))  float        v8f;
typedef __attribute__((ext_vector_type(4)))  unsigned int u32x4;

union Frag16 { v16h h; u32x4 q[2]; };

#define DEV __device__ __forceinline__

DEV void atomAddF(float* p, float v) {
  __hip_atomic_fetch_add(p, v, __ATOMIC_RELAXED, __HIP_MEMORY_SCOPE_AGENT);
}

DEV float sigm(float x) { return 1.0f / (1.0f + __expf(-x)); }

// ---------------- WMMA GEMM: C[M,Dout] = A16[M,K] @ Wt16[Dout,K]^T + bias ----------------
// grid.x = M/16, block = 32*(Dout/16), dynamic LDS = 16*K*2 bytes.
__global__ __launch_bounds__(256) void k_gemm_wmma(
    const _Float16* __restrict__ A, const _Float16* __restrict__ Wt,
    const float* __restrict__ bias, float* __restrict__ C,
    int M, int K, int Dout)
{
  extern __shared__ _Float16 As[];
  const int row0 = blockIdx.x * 16;

  // cooperative copy of contiguous 16xK A tile into LDS (b128 chunks)
  const size_t chunks = ((size_t)16 * K) / 8;     // 8 halfs per 16B
  const u32x4* srcT = (const u32x4*)(A + (size_t)row0 * K);
  u32x4* dstT = (u32x4*)As;
  for (size_t t = threadIdx.x; t < chunks; t += blockDim.x) dstT[t] = srcT[t];
  __syncthreads();

  const int wave = threadIdx.x >> 5;
  const int lane = threadIdx.x & 31;
  const int col0 = wave * 16;
  const int r    = lane & 15;
  const int hi   = lane >> 4;                     // 0 or 1

  // A frag: lane holds row r; halves 0..7 -> K = k0+kb+0..7, 8..15 -> K = k0+kb+16..23 (kb = hi*8)
  const _Float16* arow = As + (size_t)r * K + hi * 8;
  // B frag: lane holds col r; halves 0..7 -> K = k0+kb+0..7, 8..15 -> K = k0+kb+16..23 (kb = hi*8)
  const _Float16* brow = Wt + (size_t)(col0 + r) * K + hi * 8;

  v8f acc = {};
  for (int k0 = 0; k0 < K; k0 += 32) {
    Frag16 a, b;
    a.q[0] = *(const u32x4*)(arow + k0);
    a.q[1] = *(const u32x4*)(arow + k0 + 16);
    b.q[0] = *(const u32x4*)(brow + k0);
    b.q[1] = *(const u32x4*)(brow + k0 + 16);
    acc = __builtin_amdgcn_wmma_f32_16x16x32_f16(false, a.h, false, b.h,
                                                 (short)0, acc, false, false);
  }

  const int col = col0 + r;
  const int rb  = row0 + hi * 8;                  // VGPR v -> row rb+v
  const float bv = bias[col];
#pragma unroll
  for (int v = 0; v < 8; ++v)
    C[(size_t)(rb + v) * Dout + col] = acc[v] + bv;
}

// ---------------- small utility kernels ----------------
__global__ void k_cvt16(const float* __restrict__ in, _Float16* __restrict__ out, long n) {
  long i = (long)blockIdx.x * blockDim.x + threadIdx.x;
  if (i < n) out[i] = (_Float16)in[i];
}

__global__ void k_transpose16(const float* __restrict__ in, _Float16* __restrict__ out,
                              int K, int Dout) {
  int i = blockIdx.x * blockDim.x + threadIdx.x;
  if (i >= K * Dout) return;
  int k = i / Dout, d = i % Dout;
  out[(size_t)d * K + k] = (_Float16)in[i];
}

__global__ void k_pack_hp(const _Float16* __restrict__ h16, const _Float16* __restrict__ p16,
                          _Float16* __restrict__ hp16) {
  long idx = (long)blockIdx.x * blockDim.x + threadIdx.x;
  if (idx >= (long)N_ * D_) return;
  long i = idx >> 7; int d = (int)(idx & 127);
  hp16[i * 256 + d]       = h16[idx];
  hp16[i * 256 + 128 + d] = p16[idx];
}

__global__ void k_proj_p(const float* __restrict__ pos, const float* __restrict__ Wp,
                         const float* __restrict__ bp, float* __restrict__ p) {
  long idx = (long)blockIdx.x * blockDim.x + threadIdx.x;
  if (idx >= (long)N_ * D_) return;
  long i = idx >> 7; int d = (int)(idx & 127);
  float a = bp[d];
#pragma unroll
  for (int k = 0; k < P_; ++k) a += pos[i * P_ + k] * Wp[k * D_ + d];
  p[idx] = a;
}

__global__ void k_proj_e(const float* __restrict__ ef, const float* __restrict__ We,
                         const float* __restrict__ be, float* __restrict__ e) {
  long idx = (long)blockIdx.x * blockDim.x + threadIdx.x;
  if (idx >= (long)E_ * D_) return;
  long i = idx >> 7; int d = (int)(idx & 127);
  e[idx] = ef[i] * We[d] + be[d];
}

// ---------------- edge kernels ----------------
// hat (pre-loaded with B3e) += B1h[src] + B2h[dst]; ssum[dst] += sigmoid(hat)
__global__ __launch_bounds__(256) void k_edge1(
    const int* __restrict__ src, const int* __restrict__ dst,
    const float* __restrict__ B1h, const float* __restrict__ B2h,
    float* __restrict__ hat, float* __restrict__ ssum)
{
  long idx = (long)blockIdx.x * blockDim.x + threadIdx.x;
  if (idx >= (long)E_ * 32) return;
  int e = (int)(idx >> 5);
  int q = ((int)idx & 31) << 2;
  int s = src[e], d = dst[e];
  float4 hv = *(const float4*)(hat + (size_t)e * D_ + q);
  float4 u  = *(const float4*)(B1h + (size_t)s * D_ + q);
  float4 w  = *(const float4*)(B2h + (size_t)d * D_ + q);
  float4 o;
  o.x = hv.x + u.x + w.x;  o.y = hv.y + u.y + w.y;
  o.z = hv.z + u.z + w.z;  o.w = hv.w + u.w + w.w;
  *(float4*)(hat + (size_t)e * D_ + q) = o;
  float* sp = ssum + (size_t)d * D_ + q;
  atomAddF(sp + 0, sigm(o.x));
  atomAddF(sp + 1, sigm(o.y));
  atomAddF(sp + 2, sigm(o.z));
  atomAddF(sp + 3, sigm(o.w));
}

// eta = sigma/(ssum[dst]+1e-6); hagg[dst]+=eta*A2h[src]; pagg[dst]+=eta*C2p[src]
__global__ __launch_bounds__(256) void k_edge2(
    const int* __restrict__ src, const int* __restrict__ dst,
    const float* __restrict__ hat, const float* __restrict__ ssum,
    const float* __restrict__ A2h, const float* __restrict__ C2p,
    float* __restrict__ hagg, float* __restrict__ pagg)
{
  long idx = (long)blockIdx.x * blockDim.x + threadIdx.x;
  if (idx >= (long)E_ * 32) return;
  int e = (int)(idx >> 5);
  int q = ((int)idx & 31) << 2;
  int s = src[e], d = dst[e];
  float4 hv = *(const float4*)(hat  + (size_t)e * D_ + q);
  float4 sv = *(const float4*)(ssum + (size_t)d * D_ + q);
  float e0 = sigm(hv.x) / (sv.x + 1e-6f);
  float e1 = sigm(hv.y) / (sv.y + 1e-6f);
  float e2 = sigm(hv.z) / (sv.z + 1e-6f);
  float e3 = sigm(hv.w) / (sv.w + 1e-6f);
  float4 a = *(const float4*)(A2h + (size_t)s * D_ + q);
  float4 c = *(const float4*)(C2p + (size_t)s * D_ + q);
  float* hp = hagg + (size_t)d * D_ + q;
  float* pp = pagg + (size_t)d * D_ + q;
  atomAddF(hp + 0, e0 * a.x); atomAddF(hp + 1, e1 * a.y);
  atomAddF(hp + 2, e2 * a.z); atomAddF(hp + 3, e3 * a.w);
  atomAddF(pp + 0, e0 * c.x); atomAddF(pp + 1, e1 * c.y);
  atomAddF(pp + 2, e2 * c.z); atomAddF(pp + 3, e3 * c.w);
}

// ---------------- batchnorm / residual kernels ----------------
__global__ void k_node_pre(const float* __restrict__ A1h, float* __restrict__ hagg,
                           const float* __restrict__ snorm) {
  long idx = (long)blockIdx.x * blockDim.x + threadIdx.x;
  if (idx >= (long)N_ * D_) return;
  long i = idx >> 7;
  hagg[idx] = (A1h[idx] + hagg[idx]) * snorm[i];
}

__global__ __launch_bounds__(256) void k_bn_stats(
    const float* __restrict__ x, float* __restrict__ sum, float* __restrict__ sq, long rows)
{
  __shared__ float ls[256], lq[256];
  int d = threadIdx.x & 127;
  int sub = threadIdx.x >> 7;
  float s = 0.f, s2 = 0.f;
  for (long r = (long)blockIdx.x * 2 + sub; r < rows; r += (long)gridDim.x * 2) {
    float v = x[r * D_ + d];
    s += v; s2 += v * v;
  }
  ls[threadIdx.x] = s; lq[threadIdx.x] = s2;
  __syncthreads();
  if (threadIdx.x < 128) {
    atomAddF(&sum[d], ls[threadIdx.x] + ls[threadIdx.x + 128]);
    atomAddF(&sq[d],  lq[threadIdx.x] + lq[threadIdx.x + 128]);
  }
}

// y += relu(g*(x-m)/sqrt(v+eps)+b), also writes f16 copy of y
__global__ void k_bn_apply_res(float* __restrict__ y, _Float16* __restrict__ y16,
                               const float* __restrict__ x,
                               const float* __restrict__ sum, const float* __restrict__ sq,
                               const float* __restrict__ g, const float* __restrict__ b,
                               float invRows, long n)
{
  long idx = (long)blockIdx.x * blockDim.x + threadIdx.x;
  if (idx >= n) return;
  int d = (int)(idx & 127);
  float m = sum[d] * invRows;
  float var = sq[d] * invRows - m * m;
  float t = g[d] * (x[idx] - m) * rsqrtf(var + 1e-5f) + b[d];
  t = fmaxf(t, 0.f);
  float yv = y[idx] + t;
  y[idx] = yv;
  y16[idx] = (_Float16)yv;
}

__global__ void k_p_update(float* __restrict__ p, _Float16* __restrict__ p16,
                           const float* __restrict__ C1p, const float* __restrict__ pagg) {
  long idx = (long)blockIdx.x * blockDim.x + threadIdx.x;
  if (idx >= (long)N_ * D_) return;
  float pv = p[idx] + tanhf(C1p[idx] + pagg[idx]);
  p[idx] = pv;
  p16[idx] = (_Float16)pv;
}

// ---------------- readout ----------------
__global__ void k_gid(const int* __restrict__ bnn, int* __restrict__ gid) {
  int i = blockIdx.x * blockDim.x + threadIdx.x;
  if (i >= N_) return;
  int acc = 0, gg = G_ - 1;
  for (int g = 0; g < G_; ++g) {
    int nacc = acc + bnn[g];
    if (i < nacc) { gg = g; break; }
    acc = nacc;
  }
  gid[i] = gg;
}

__global__ void k_gsum(const float* __restrict__ p2, const int* __restrict__ gid,
                       float* __restrict__ gsum) {
  long idx = (long)blockIdx.x * blockDim.x + threadIdx.x;
  if (idx >= (long)N_ * P_) return;
  long i = idx >> 4; int d = (int)(idx & 15);
  atomAddF(&gsum[gid[i] * P_ + d], p2[idx]);
}

__global__ void k_gss(const float* __restrict__ p2, const int* __restrict__ gid,
                      const int* __restrict__ bnn, const float* __restrict__ gsum,
                      float* __restrict__ gss) {
  long idx = (long)blockIdx.x * blockDim.x + threadIdx.x;
  if (idx >= (long)N_ * P_) return;
  long i = idx >> 4; int d = (int)(idx & 15);
  int g = gid[i];
  float c = p2[idx] - gsum[g * P_ + d] / (float)bnn[g];
  atomAddF(&gss[g * P_ + d], c * c);
}

__global__ __launch_bounds__(128) void k_readout(
    const float* __restrict__ h, const float* __restrict__ p2,
    const float* __restrict__ gsum, const float* __restrict__ gss,
    const int* __restrict__ bnn,
    const float* __restrict__ WhpW, const float* __restrict__ Whpb,
    const float* __restrict__ W0, const float* __restrict__ b0,
    const float* __restrict__ W1, const float* __restrict__ b1,
    const float* __restrict__ W2, const float* __restrict__ b2,
    float* __restrict__ out)
{
  __shared__ float pf[16], hpv[128], y0[64], y1[32];
  int g = blockIdx.x, t = threadIdx.x;
  int node = 0;
  for (int q = 0; q < g; ++q) node += bnn[q];
  if (t < 16) {
    float cnt = (float)bnn[g];
    float c = p2[(size_t)node * P_ + t] - gsum[g * P_ + t] / cnt;
    pf[t] = c / sqrtf(gss[g * P_ + t]);
  }
  __syncthreads();
  float acc = Whpb[t];
  for (int k = 0; k < 128; ++k) acc += h[(size_t)node * D_ + k] * WhpW[k * D_ + t];
  for (int k = 0; k < 16; ++k)  acc += pf[k] * WhpW[(128 + k) * D_ + t];
  hpv[t] = acc;
  __syncthreads();
  if (t < 64) { float a = b0[t]; for (int k = 0; k < 128; ++k) a += hpv[k] * W0[k * 64 + t]; y0[t] = fmaxf(a, 0.f); }
  __syncthreads();
  if (t < 32) { float a = b1[t]; for (int k = 0; k < 64; ++k) a += y0[k] * W1[k * 32 + t]; y1[t] = fmaxf(a, 0.f); }
  __syncthreads();
  if (t == 0) { float a = b2[0]; for (int k = 0; k < 32; ++k) a += y1[k] * W2[k]; out[g] = a; }
}

// ---------------- host side ----------------
static inline void gemm(const _Float16* A, const _Float16* Wt, const float* bias, float* C,
                        int M, int K, int Dout, hipStream_t s) {
  k_gemm_wmma<<<M / 16, 32 * (Dout / 16), (size_t)16 * K * 2, s>>>(A, Wt, bias, C, M, K, Dout);
}

extern "C" void kernel_launch(void* const* d_in, const int* in_sizes, int n_in,
                              void* d_out, int out_size, void* d_ws, size_t ws_size,
                              hipStream_t stream)
{
  const float* node_feat = (const float*)d_in[0];
  const float* pos_enc   = (const float*)d_in[1];
  const float* edge_feat = (const float*)d_in[2];
  const float* snorm     = (const float*)d_in[3];
  const int*   esrc      = (const int*)d_in[4];
  const int*   edst      = (const int*)d_in[5];
  const int*   bnn       = (const int*)d_in[6];
  const float* W_h  = (const float*)d_in[7];
  const float* b_h  = (const float*)d_in[8];
  const float* W_p  = (const float*)d_in[9];
  const float* b_p  = (const float*)d_in[10];
  const float* W_e  = (const float*)d_in[11];
  const float* b_e  = (const float*)d_in[12];
  const float* A1W  = (const float*)d_in[13];
  const float* A1b  = (const float*)d_in[14];
  const float* A2W  = (const float*)d_in[15];
  const float* A2b  = (const float*)d_in[16];
  const float* B1W  = (const float*)d_in[17];
  const float* B1b  = (const float*)d_in[18];
  const float* B2W  = (const float*)d_in[19];
  const float* B2b  = (const float*)d_in[20];
  const float* B3W  = (const float*)d_in[21];
  const float* B3b  = (const float*)d_in[22];
  const float* C1W  = (const float*)d_in[23];
  const float* C1b  = (const float*)d_in[24];
  const float* C2W  = (const float*)d_in[25];
  const float* C2b  = (const float*)d_in[26];
  const float* bnhg = (const float*)d_in[27];
  const float* bnhb = (const float*)d_in[28];
  const float* bneg = (const float*)d_in[29];
  const float* bneb = (const float*)d_in[30];
  const float* poutW = (const float*)d_in[31];
  const float* poutb = (const float*)d_in[32];
  const float* WhpW  = (const float*)d_in[33];
  const float* Whpb  = (const float*)d_in[34];
  const float* W0 = (const float*)d_in[35];
  const float* b0 = (const float*)d_in[36];
  const float* W1 = (const float*)d_in[37];
  const float* b1 = (const float*)d_in[38];
  const float* W2 = (const float*)d_in[39];
  const float* b2 = (const float*)d_in[40];
  float* out = (float*)d_out;

  char* base = (char*)d_ws;
  size_t off = 0;
  auto take = [&](size_t bytes) -> char* {
    char* pp = base + off;
    off += (bytes + 255) & ~(size_t)255;
    return pp;
  };

  const size_t ND4 = (size_t)N_ * D_ * 4;
  float*    h    = (float*)take(ND4);
  float*    p    = (float*)take(ND4);
  _Float16* h16  = (_Float16*)take((size_t)N_ * D_ * 2);
  _Float16* p16  = (_Float16*)take((size_t)N_ * D_ * 2);
  _Float16* hp16 = (_Float16*)take((size_t)N_ * 2 * D_ * 2);
  _Float16* nf16 = (_Float16*)take((size_t)N_ * FIN_ * 2);
  float*    e    = (float*)take((size_t)E_ * D_ * 4);
  _Float16* e16  = (_Float16*)take((size_t)E_ * D_ * 2);
  float*    hat  = (float*)take((size_t)E_ * D_ * 4);
  float* A1h  = (float*)take(ND4);
  float* A2h  = (float*)take(ND4);
  float* B1h  = (float*)take(ND4);
  float* B2h  = (float*)take(ND4);
  float* C1p  = (float*)take(ND4);
  float* C2p  = (float*)take(ND4);
  float* ssum = (float*)take(ND4);
  float* hagg = (float*)take(ND4);
  float* pagg = (float*)take(ND4);
  float* p2   = (float*)take((size_t)N_ * P_ * 4);
  int*   gid  = (int*)take((size_t)N_ * 4);
  float* bnsum = (float*)take(256 * 4);     float* bnsq = bnsum + 128;
  float* gsum  = (float*)take(2 * G_ * P_ * 4); float* gss = gsum + G_ * P_;
  _Float16* Wht   = (_Float16*)take((size_t)128 * 64 * 2);
  _Float16* A1t   = (_Float16*)take((size_t)3 * 128 * 256 * 2);
  _Float16* A2t   = (_Float16*)take((size_t)3 * 128 * 256 * 2);
  _Float16* B1t   = (_Float16*)take((size_t)3 * 128 * 128 * 2);
  _Float16* B2t   = (_Float16*)take((size_t)3 * 128 * 128 * 2);
  _Float16* B3t   = (_Float16*)take((size_t)3 * 128 * 128 * 2);
  _Float16* C1t   = (_Float16*)take((size_t)3 * 128 * 128 * 2);
  _Float16* C2t   = (_Float16*)take((size_t)3 * 128 * 128 * 2);
  _Float16* poutt = (_Float16*)take((size_t)16 * 128 * 2);

  auto tr = [&](const float* in, _Float16* o, int K, int Dout) {
    int n = K * Dout;
    k_transpose16<<<(n + 255) / 256, 256, 0, stream>>>(in, o, K, Dout);
  };

  // weight prep (f16 transposed [Dout,K])
  tr(W_h, Wht, 64, 128);
  for (int l = 0; l < L_; ++l) {
    tr(A1W + (size_t)l * 256 * 128, A1t + (size_t)l * 128 * 256, 256, 128);
    tr(A2W + (size_t)l * 256 * 128, A2t + (size_t)l * 128 * 256, 256, 128);
    tr(B1W + (size_t)l * 128 * 128, B1t + (size_t)l * 128 * 128, 128, 128);
    tr(B2W + (size_t)l * 128 * 128, B2t + (size_t)l * 128 * 128, 128, 128);
    tr(B3W + (size_t)l * 128 * 128, B3t + (size_t)l * 128 * 128, 128, 128);
    tr(C1W + (size_t)l * 128 * 128, C1t + (size_t)l * 128 * 128, 128, 128);
    tr(C2W + (size_t)l * 128 * 128, C2t + (size_t)l * 128 * 128, 128, 128);
  }
  tr(poutW, poutt, 128, 16);

  const long nND = (long)N_ * D_;
  const long nED = (long)E_ * D_;
  const int  bND = (int)((nND + 255) / 256);
  const int  bED = (int)((nED + 255) / 256);

  // input projections
  k_cvt16<<<(int)(((long)N_ * FIN_ + 255) / 256), 256, 0, stream>>>(node_feat, nf16, (long)N_ * FIN_);
  gemm(nf16, Wht, b_h, h, N_, 64, 128, stream);                 // h = node_feat @ W_h + b_h
  k_proj_p<<<bND, 256, 0, stream>>>(pos_enc, W_p, b_p, p);      // p = pos_enc @ W_p + b_p
  k_proj_e<<<bED, 256, 0, stream>>>(edge_feat, W_e, b_e, e);    // e = edge_feat @ W_e + b_e
  k_cvt16<<<bND, 256, 0, stream>>>(h, h16, nND);
  k_cvt16<<<bND, 256, 0, stream>>>(p, p16, nND);
  k_cvt16<<<bED, 256, 0, stream>>>(e, e16, nED);

  const int bE32 = (int)(((long)E_ * 32 + 255) / 256);

  for (int l = 0; l < L_; ++l) {
    k_pack_hp<<<bND, 256, 0, stream>>>(h16, p16, hp16);
    gemm(hp16, A1t + (size_t)l * 128 * 256, A1b + l * 128, A1h, N_, 256, 128, stream);
    gemm(hp16, A2t + (size_t)l * 128 * 256, A2b + l * 128, A2h, N_, 256, 128, stream);
    gemm(h16,  B1t + (size_t)l * 128 * 128, B1b + l * 128, B1h, N_, 128, 128, stream);
    gemm(h16,  B2t + (size_t)l * 128 * 128, B2b + l * 128, B2h, N_, 128, 128, stream);
    gemm(p16,  C1t + (size_t)l * 128 * 128, C1b + l * 128, C1p, N_, 128, 128, stream);
    gemm(p16,  C2t + (size_t)l * 128 * 128, C2b + l * 128, C2p, N_, 128, 128, stream);
    gemm(e16,  B3t + (size_t)l * 128 * 128, B3b + l * 128, hat, E_, 128, 128, stream); // hat = B3e

    (void)hipMemsetAsync(ssum, 0, ND4, stream);
    k_edge1<<<bE32, 256, 0, stream>>>(esrc, edst, B1h, B2h, hat, ssum);

    (void)hipMemsetAsync(hagg, 0, ND4, stream);
    (void)hipMemsetAsync(pagg, 0, ND4, stream);
    k_edge2<<<bE32, 256, 0, stream>>>(esrc, edst, hat, ssum, A2h, C2p, hagg, pagg);

    k_node_pre<<<bND, 256, 0, stream>>>(A1h, hagg, snorm);      // hagg = (A1h+hagg)*snorm
    (void)hipMemsetAsync(bnsum, 0, 256 * 4, stream);
    k_bn_stats<<<2048, 256, 0, stream>>>(hagg, bnsum, bnsq, (long)N_);
    k_bn_apply_res<<<bND, 256, 0, stream>>>(h, h16, hagg, bnsum, bnsq,
                                            bnhg + l * 128, bnhb + l * 128,
                                            1.0f / (float)N_, nND);
    k_p_update<<<bND, 256, 0, stream>>>(p, p16, C1p, pagg);

    if (l < L_ - 1) {   // e is dead after the last layer
      (void)hipMemsetAsync(bnsum, 0, 256 * 4, stream);
      k_bn_stats<<<2048, 256, 0, stream>>>(hat, bnsum, bnsq, (long)E_);
      k_bn_apply_res<<<bED, 256, 0, stream>>>(e, e16, hat, bnsum, bnsq,
                                              bneg + l * 128, bneb + l * 128,
                                              1.0f / (float)E_, nED);
    }
  }

  // readout
  gemm(p16, poutt, poutb, p2, N_, 128, 16, stream);             // p2 = p @ pout_W + b
  k_gid<<<(N_ + 255) / 256, 256, 0, stream>>>(bnn, gid);
  (void)hipMemsetAsync(gsum, 0, 2 * G_ * P_ * 4, stream);
  const int bNP = (int)(((long)N_ * P_ + 255) / 256);
  k_gsum<<<bNP, 256, 0, stream>>>(p2, gid, gsum);
  k_gss<<<bNP, 256, 0, stream>>>(p2, gid, bnn, gsum, gss);
  k_readout<<<G_, 128, 0, stream>>>(h, p2, gsum, gss, bnn, WhpW, Whpb,
                                    W0, b0, W1, b1, W2, b2, out);
  (void)in_sizes; (void)n_in; (void)out_size; (void)ws_size;
}
